// GPTv2Base_64570538328792
// MI455X (gfx1250) — compile-verified, matrix operand order checked by reading, not run on hardware
//
#include <hip/hip_runtime.h>
#include <hip/hip_bf16.h>

// ---------------------------------------------------------------------------
// Types for CDNA5 WMMA (wave32): v16bf A/B operands, v8f accumulator.
// ---------------------------------------------------------------------------
typedef __bf16 bf16;
typedef __attribute__((ext_vector_type(16))) __bf16 v16bf;
typedef __attribute__((ext_vector_type(8)))  float  v8f;

union FragBF {
    uint4 u[2];
    v16bf v;
};

// Model constants
#define BB 4
#define TT 1024
#define EE 1024
#define HH 16
#define DD 64
#define LL 6
#define FFF 4096
#define VV 32000
#define MM (BB * TT)   // 4096 rows

// ---------------------------------------------------------------------------
// CDNA5 async global->LDS copy (ASYNCcnt-tracked DMA, no VGPR bounce).
// Generic LDS pointers hold the LDS byte offset in their low 32 bits.
// ---------------------------------------------------------------------------
#if defined(__gfx1250__)
__device__ __forceinline__ void async_copy_b128(void* lds, const void* gaddr) {
    asm volatile("global_load_async_to_lds_b128 %0, %1, off"
                 :: "v"((unsigned)(uintptr_t)lds),
                    "v"((unsigned long long)(uintptr_t)gaddr)
                 : "memory");
}
__device__ __forceinline__ void wait_async() {
    asm volatile("s_wait_asynccnt 0x0" ::: "memory");
}
#else
__device__ __forceinline__ void async_copy_b128(void* lds, const void* gaddr) {
    *(uint4*)lds = *(const uint4*)gaddr;
}
__device__ __forceinline__ void wait_async() {}
#endif

// ---------------------------------------------------------------------------
// f32 -> bf16 conversion (weights staging)
// ---------------------------------------------------------------------------
__global__ void f32_to_bf16_kernel(const float* __restrict__ in,
                                   bf16* __restrict__ out, size_t n) {
    size_t i = (size_t)blockIdx.x * blockDim.x + threadIdx.x;
    size_t stride = (size_t)gridDim.x * blockDim.x;
    for (; i < n; i += stride) out[i] = (bf16)in[i];
}

// ---------------------------------------------------------------------------
// Embedding: x[b,t,:] = tok_emb[idx[b,t],:] + pos_emb[t,:]
// ---------------------------------------------------------------------------
__global__ void embed_kernel(const int* __restrict__ idx,
                             const float* __restrict__ tok,
                             const float* __restrict__ pos,
                             float* __restrict__ x, size_t n) {
    size_t i = (size_t)blockIdx.x * blockDim.x + threadIdx.x;
    size_t stride = (size_t)gridDim.x * blockDim.x;
    for (; i < n; i += stride) {
        size_t bt = i / EE;
        size_t e  = i % EE;
        size_t t  = bt % TT;
        x[i] = tok[(size_t)idx[bt] * EE + e] + pos[t * EE + e];
    }
}

// ---------------------------------------------------------------------------
// Row LayerNorm: f32 in -> bf16 out (feeds WMMA GEMM A operand)
// ---------------------------------------------------------------------------
__global__ __launch_bounds__(256)
void layernorm_kernel(const float* __restrict__ x,
                      const float* __restrict__ sc,
                      const float* __restrict__ bi,
                      bf16* __restrict__ out) {
    __shared__ float red[256];
    const int row = blockIdx.x;
    const float* xr = x + (size_t)row * EE;
    float sum = 0.f, sq = 0.f;
    for (int i = threadIdx.x; i < EE; i += 256) {
        float v = xr[i];
        sum += v;
        sq  += v * v;
    }
    red[threadIdx.x] = sum;
    __syncthreads();
    for (int s = 128; s > 0; s >>= 1) {
        if (threadIdx.x < s) red[threadIdx.x] += red[threadIdx.x + s];
        __syncthreads();
    }
    float mu = red[0] / EE;
    __syncthreads();
    red[threadIdx.x] = sq;
    __syncthreads();
    for (int s = 128; s > 0; s >>= 1) {
        if (threadIdx.x < s) red[threadIdx.x] += red[threadIdx.x + s];
        __syncthreads();
    }
    float var = red[0] / EE - mu * mu;
    float r = rsqrtf(var + 1e-5f);
    bf16* orow = out + (size_t)row * EE;
    for (int i = threadIdx.x; i < EE; i += 256)
        orow[i] = (bf16)((xr[i] - mu) * r * sc[i] + bi[i]);
}

// ---------------------------------------------------------------------------
// bf16 WMMA GEMM:  C[M,N] = A[M,K] * B[K,N] (+bias) (+residual) (relu?)
//   Tiles: 128x64x32. 256 threads = 8 waves; wave w owns rows w*16..w*16+15
//   and sweeps 4 column subtiles of 16 -> 4x v_wmma_f32_16x16x32_bf16 / ktile.
//   A tile is staged with global_load_async_to_lds_b128 (ASYNCcnt DMA);
//   B tile is transposed through registers into Bs[n][k].
// WMMA 16-bit operand layout (ISA 7.12.2): per-lane element e maps to
//   K = (e<8 ? e : e+8) + (lane>=16 ? 8 : 0); lane%16 = row (A) / col (B);
//   C/D element i -> row i + (lane>=16)*8, col lane%16.
// ---------------------------------------------------------------------------
__global__ __launch_bounds__(256)
void gemm_bf16_kernel(const bf16* __restrict__ A, const bf16* __restrict__ B,
                      const float* __restrict__ bias,
                      const float* __restrict__ resid,
                      float* __restrict__ Cf, bf16* __restrict__ Cb,
                      int M, int N, int K, int ldb, int ldc,
                      long long strideB, long long strideC, int relu) {
    __shared__ bf16 As[128 * 32];   // [row][k]
    __shared__ bf16 Bs[64 * 32];    // transposed: [n][k]
    const int tid  = threadIdx.x;
    const int wav  = tid >> 5;
    const int lane = tid & 31;
    const int l16  = lane & 15;
    const int ksel = (lane < 16) ? 0 : 8;
    const int m0   = blockIdx.y * 128;
    const int n0   = blockIdx.x * 64;
    const long long z = blockIdx.z;
    B += z * strideB;
    if (Cf)    Cf    += z * strideC;
    if (Cb)    Cb    += z * strideC;
    if (resid) resid += z * strideC;

    v8f acc[4] = {};
    for (int k0 = 0; k0 < K; k0 += 32) {
        // A tile: 128x32 bf16 = 512 16-byte chunks; async DMA to LDS
        for (int q = tid; q < 512; q += 256) {
            int row = q >> 2;
            int kp  = (q & 3) << 3;
            const bf16* gA = &A[(size_t)(m0 + row) * K + k0 + kp];
            async_copy_b128(&As[row * 32 + kp], gA);
            if (k0 + 32 < K) __builtin_prefetch(gA + 32, 0, 0);
        }
        // B tile: 32x64 bf16, transposed into Bs[n][k]
        {
            int q  = tid;               // 256 uint4
            int br = q >> 3;            // k row 0..31
            int np = (q & 7) << 3;      // n 0..56
            const bf16* gB = &B[(size_t)(k0 + br) * ldb + n0 + np];
            uint4 t = *(const uint4*)gB;
            if (k0 + 32 < K)
                __builtin_prefetch(&B[(size_t)(k0 + 32 + br) * ldb + n0 + np], 0, 0);
            const bf16* e = (const bf16*)&t;
#pragma unroll
            for (int j = 0; j < 8; ++j) Bs[(np + j) * 32 + br] = e[j];
        }
        wait_async();
        __syncthreads();
        FragBF a;
        {
            const bf16* p = &As[(wav * 16 + l16) * 32 + ksel];
            a.u[0] = *(const uint4*)p;
            a.u[1] = *(const uint4*)(p + 16);
        }
#pragma unroll
        for (int c = 0; c < 4; ++c) {
            FragBF b;
            const bf16* p = &Bs[(c * 16 + l16) * 32 + ksel];
            b.u[0] = *(const uint4*)p;
            b.u[1] = *(const uint4*)(p + 16);
            acc[c] = __builtin_amdgcn_wmma_f32_16x16x32_bf16(
                false, a.v, false, b.v, (short)0, acc[c], false, false);
        }
        __syncthreads();
    }
    // Epilogue
#pragma unroll
    for (int c = 0; c < 4; ++c) {
        int gn = n0 + c * 16 + l16;
        float bv = bias ? bias[gn] : 0.0f;
#pragma unroll
        for (int i = 0; i < 8; ++i) {
            int gm = m0 + wav * 16 + i + ((lane >= 16) ? 8 : 0);
            size_t off = (size_t)gm * ldc + gn;
            float v = acc[c][i] + bv;
            if (resid) v += resid[off];
            if (relu)  v = fmaxf(v, 0.0f);
            if (Cf) Cf[off] = v;
            else    Cb[off] = (bf16)v;
        }
    }
}

// ---------------------------------------------------------------------------
// Flash attention (causal, scale = E^-0.5 = 1/32 as in the reference).
// One wave (32 threads) per 16 query rows per (b,h). Q,K,V,O are bf16 in
// [B,T,H*D] layout (head h occupies columns h*64..h*64+63).
// Scores via WMMA QK^T (K-dim 64 -> 2 k-steps), P*V via WMMA after bouncing
// P through LDS to re-shape C-layout -> A-layout.
// ---------------------------------------------------------------------------
__global__ __launch_bounds__(32)
void attn_kernel(const bf16* __restrict__ Q, const bf16* __restrict__ Kb,
                 const bf16* __restrict__ Vb, bf16* __restrict__ O,
                 float scale) {
    __shared__ bf16 Pls[16 * 32];   // P tile, row-major [16][32]
    __shared__ bf16 Vt[64 * 32];    // V tile transposed: [d][key]
    const int lane = threadIdx.x;
    const int l16  = lane & 15;
    const int hlf  = lane >> 4;
    const int ksel = hlf ? 8 : 0;
    const int qbase = blockIdx.x * 16;
    const int bh = blockIdx.y;
    const int b  = bh / HH;
    const int h  = bh % HH;
    const size_t base = (size_t)b * TT * EE + (size_t)h * DD;   // + t*EE

    // Q fragments (A operand), K-dim steps 0..31, 32..63
    FragBF qf[2];
    {
        const bf16* qrow = Q + base + (size_t)(qbase + l16) * EE;
#pragma unroll
        for (int s = 0; s < 2; ++s) {
            qf[s].u[0] = *(const uint4*)(qrow + s * 32 + ksel);
            qf[s].u[1] = *(const uint4*)(qrow + s * 32 + ksel + 16);
        }
    }
    v8f accO[4] = {};
    float mrow[8], lrow[8];
#pragma unroll
    for (int i = 0; i < 8; ++i) { mrow[i] = -1e30f; lrow[i] = 0.f; }

    const int jend = qbase + 16;                  // causal bound
    for (int j = 0; j < jend; j += 32) {
        // Stage V[j..j+31][0..63] transposed into Vt[d][key]
        {
            const bf16* vrow = Vb + base + (size_t)(j + lane) * EE;
            if (j + 32 < jend) {
                __builtin_prefetch(vrow + (size_t)32 * EE, 0, 0);
                __builtin_prefetch(Kb + base + (size_t)(j + 32 + lane) * EE, 0, 0);
            }
#pragma unroll
            for (int u = 0; u < 8; ++u) {
                uint4 t = *(const uint4*)(vrow + u * 8);
                const bf16* e = (const bf16*)&t;
#pragma unroll
                for (int x2 = 0; x2 < 8; ++x2)
                    Vt[(u * 8 + x2) * 32 + lane] = e[x2];
            }
        }
        __syncthreads();

        // Scores S[sub] = Q * K^T for key subtiles [j..j+15], [j+16..j+31]
        v8f S[2];
#pragma unroll
        for (int sub = 0; sub < 2; ++sub) {
            v8f s = {};
#pragma unroll
            for (int kd = 0; kd < 2; ++kd) {
                FragBF kf;
                const bf16* krow =
                    Kb + base + (size_t)(j + sub * 16 + l16) * EE + kd * 32;
                kf.u[0] = *(const uint4*)(krow + ksel);
                kf.u[1] = *(const uint4*)(krow + ksel + 16);
                s = __builtin_amdgcn_wmma_f32_16x16x32_bf16(
                    false, qf[kd].v, false, kf.v, (short)0, s, false, false);
            }
            S[sub] = s;
        }

        // Online softmax update (rows live in half-waves; xor-reduce over 16)
#pragma unroll
        for (int i = 0; i < 8; ++i) {
            int qg = qbase + i + hlf * 8;
            float s0 = S[0][i] * scale;
            float s1 = S[1][i] * scale;
            if (j + l16 > qg)      s0 = -1e30f;
            if (j + 16 + l16 > qg) s1 = -1e30f;
            float mx = fmaxf(s0, s1);
            mx = fmaxf(mx, __shfl_xor(mx, 1, 32));
            mx = fmaxf(mx, __shfl_xor(mx, 2, 32));
            mx = fmaxf(mx, __shfl_xor(mx, 4, 32));
            mx = fmaxf(mx, __shfl_xor(mx, 8, 32));
            float mnew = fmaxf(mrow[i], mx);
            float al = __expf(mrow[i] - mnew);
            float p0 = __expf(s0 - mnew);
            float p1 = __expf(s1 - mnew);
            float ps = p0 + p1;
            ps += __shfl_xor(ps, 1, 32);
            ps += __shfl_xor(ps, 2, 32);
            ps += __shfl_xor(ps, 4, 32);
            ps += __shfl_xor(ps, 8, 32);
            lrow[i] = lrow[i] * al + ps;
            mrow[i] = mnew;
#pragma unroll
            for (int c = 0; c < 4; ++c) accO[c][i] *= al;
            Pls[(i + hlf * 8) * 32 + l16]      = (bf16)p0;
            Pls[(i + hlf * 8) * 32 + 16 + l16] = (bf16)p1;
        }
        __syncthreads();

        // accO += P(16x32) * V(32x64)
        FragBF pf;
        {
            const bf16* p = &Pls[l16 * 32 + ksel];
            pf.u[0] = *(const uint4*)p;
            pf.u[1] = *(const uint4*)(p + 16);
        }
#pragma unroll
        for (int c = 0; c < 4; ++c) {
            FragBF vf;
            const bf16* p = &Vt[(c * 16 + l16) * 32 + ksel];
            vf.u[0] = *(const uint4*)p;
            vf.u[1] = *(const uint4*)(p + 16);
            accO[c] = __builtin_amdgcn_wmma_f32_16x16x32_bf16(
                false, pf.v, false, vf.v, (short)0, accO[c], false, false);
        }
        __syncthreads();
    }
    // O = accO / l
#pragma unroll
    for (int c = 0; c < 4; ++c)
#pragma unroll
        for (int i = 0; i < 8; ++i) {
            int trow = qbase + i + hlf * 8;
            O[base + (size_t)trow * EE + c * 16 + l16] =
                (bf16)(accO[c][i] / lrow[i]);
        }
}

// ---------------------------------------------------------------------------
// Weighted cross-entropy: per-row lse, then deterministic two-pass mean.
// ---------------------------------------------------------------------------
__global__ __launch_bounds__(256)
void loss_row_kernel(const float* __restrict__ logits,
                     const int* __restrict__ targets,
                     float* __restrict__ rowloss) {
    __shared__ float red[256];
    const int m = blockIdx.x;
    const float* lr = logits + (size_t)m * VV;
    float mx = -1e30f;
    for (int i = threadIdx.x; i < VV; i += 256) mx = fmaxf(mx, lr[i]);
    red[threadIdx.x] = mx;
    __syncthreads();
    for (int s = 128; s > 0; s >>= 1) {
        if (threadIdx.x < s)
            red[threadIdx.x] = fmaxf(red[threadIdx.x], red[threadIdx.x + s]);
        __syncthreads();
    }
    mx = red[0];
    __syncthreads();
    float se = 0.f;
    for (int i = threadIdx.x; i < VV; i += 256) se += __expf(lr[i] - mx);
    red[threadIdx.x] = se;
    __syncthreads();
    for (int s = 128; s > 0; s >>= 1) {
        if (threadIdx.x < s) red[threadIdx.x] += red[threadIdx.x + s];
        __syncthreads();
    }
    if (threadIdx.x == 0) {
        float lse = mx + __logf(red[0]);
        int t = targets[m];
        float ce = lse - lr[t];
        float w = 1.0f;
        if (t == 4 || t == 3 || t == 1) w = 1.5f;   // CTX | QUES | SQL
        if (t == 2) w = 2.0f;                       // EOS
        if (t == 0) w = 0.1f;                       // PAD
        rowloss[m] = ce * w;
    }
}

__global__ __launch_bounds__(256)
void loss_reduce_kernel(const float* __restrict__ rowloss,
                        float* __restrict__ out, int n) {
    __shared__ float red[256];
    float s = 0.f;
    for (int i = threadIdx.x; i < n; i += 256) s += rowloss[i];
    red[threadIdx.x] = s;
    __syncthreads();
    for (int k = 128; k > 0; k >>= 1) {
        if (threadIdx.x < k) red[threadIdx.x] += red[threadIdx.x + k];
        __syncthreads();
    }
    if (threadIdx.x == 0) *out = red[0] / (float)n;
}

// ---------------------------------------------------------------------------
// Host-side orchestration
// ---------------------------------------------------------------------------
static inline void conv_bf16(hipStream_t st, const float* in, bf16* out, size_t n) {
    size_t g = (n + 255) / 256;
    if (g > 8192) g = 8192;
    f32_to_bf16_kernel<<<dim3((unsigned)g), dim3(256), 0, st>>>(in, out, n);
}

static inline void gemm(hipStream_t st, const bf16* A, const bf16* B,
                        const float* bias, const float* resid,
                        float* Cf, bf16* Cb,
                        int M, int N, int K, int ldb, int ldc,
                        int batch, long long sB, long long sC, int relu) {
    dim3 grid(N / 64, M / 128, batch), blk(256);
    gemm_bf16_kernel<<<grid, blk, 0, st>>>(A, B, bias, resid, Cf, Cb,
                                           M, N, K, ldb, ldc, sB, sC, relu);
}

extern "C" void kernel_launch(void* const* d_in, const int* in_sizes, int n_in,
                              void* d_out, int out_size, void* d_ws, size_t ws_size,
                              hipStream_t stream) {
    (void)in_sizes; (void)n_in; (void)out_size; (void)ws_size;
    const int*   idx     = (const int*)d_in[0];
    const int*   targets = (const int*)d_in[1];
    const float* tok_emb = (const float*)d_in[2];
    const float* pos_emb = (const float*)d_in[3];
    const float* Wq      = (const float*)d_in[4];
    const float* Wk      = (const float*)d_in[5];
    const float* Wv      = (const float*)d_in[6];
    const float* Wproj   = (const float*)d_in[7];
    const float* bproj   = (const float*)d_in[8];
    const float* W1      = (const float*)d_in[9];
    const float* b1      = (const float*)d_in[10];
    const float* W2      = (const float*)d_in[11];
    const float* b2w     = (const float*)d_in[12];
    const float* ln1_s   = (const float*)d_in[13];
    const float* ln1_b   = (const float*)d_in[14];
    const float* ln2_s   = (const float*)d_in[15];
    const float* ln2_b   = (const float*)d_in[16];
    const float* lnf_s   = (const float*)d_in[17];
    const float* lnf_b   = (const float*)d_in[18];
    const float* Whead   = (const float*)d_in[19];
    const float* bhead   = (const float*)d_in[20];

    // Workspace layout (all offsets 256-aligned)
    char* ws = (char*)d_ws;
    float* x      = (float*)(ws + 0);                     // 16 MB  f32 [M,E]
    bf16*  hbuf   = (bf16*) (ws + 16777216);              //  8 MB  bf16 [M,E]
    bf16*  qb     = (bf16*) (ws + 25165824);              //  8 MB
    bf16*  kb     = (bf16*) (ws + 33554432);              //  8 MB
    bf16*  vb     = (bf16*) (ws + 41943040);              //  8 MB
    bf16*  ob     = (bf16*) (ws + 50331648);              //  8 MB
    bf16*  act4   = (bf16*) (ws + 58720256);              // 32 MB  bf16 [M,FF]
    bf16*  wbuf   = (bf16*) (ws + 92274688);              // 64 MB  weight stage
    float* rloss  = (float*)(ws + 157810688);             // 16 KB

    float* logits = (float*)d_out;
    float* lossp  = logits + (size_t)MM * VV;

    // 1. Embedding
    {
        size_t n = (size_t)MM * EE;
        embed_kernel<<<dim3(8192), dim3(256), 0, stream>>>(idx, tok_emb, pos_emb, x, n);
    }

    const float scale = 1.0f / 32.0f;   // E^-0.5, reference scales by full E
    for (int l = 0; l < LL; ++l) {
        // ln1
        layernorm_kernel<<<dim3(MM), dim3(256), 0, stream>>>(
            x, ln1_s + (size_t)l * EE, ln1_b + (size_t)l * EE, hbuf);
        // QKV: batched over heads (N=64 per head, B stride E*D, C col offset 64)
        const size_t wqkv = (size_t)HH * EE * DD;
        conv_bf16(stream, Wq + (size_t)l * wqkv, wbuf, wqkv);
        gemm(stream, hbuf, wbuf, nullptr, nullptr, nullptr, qb,
             MM, 64, EE, DD, EE, HH, (long long)EE * DD, 64, 0);
        conv_bf16(stream, Wk + (size_t)l * wqkv, wbuf, wqkv);
        gemm(stream, hbuf, wbuf, nullptr, nullptr, nullptr, kb,
             MM, 64, EE, DD, EE, HH, (long long)EE * DD, 64, 0);
        conv_bf16(stream, Wv + (size_t)l * wqkv, wbuf, wqkv);
        gemm(stream, hbuf, wbuf, nullptr, nullptr, nullptr, vb,
             MM, 64, EE, DD, EE, HH, (long long)EE * DD, 64, 0);
        // Flash attention
        attn_kernel<<<dim3(TT / 16, BB * HH), dim3(32), 0, stream>>>(
            qb, kb, vb, ob, scale);
        // Output projection + residual -> x
        conv_bf16(stream, Wproj + (size_t)l * EE * EE, wbuf, (size_t)EE * EE);
        gemm(stream, ob, wbuf, bproj + (size_t)l * EE, x, x, nullptr,
             MM, EE, EE, EE, EE, 1, 0, 0, 0);
        // ln2
        layernorm_kernel<<<dim3(MM), dim3(256), 0, stream>>>(
            x, ln2_s + (size_t)l * EE, ln2_b + (size_t)l * EE, hbuf);
        // MLP fc1 (relu, bf16 out)
        conv_bf16(stream, W1 + (size_t)l * EE * FFF, wbuf, (size_t)EE * FFF);
        gemm(stream, hbuf, wbuf, b1 + (size_t)l * FFF, nullptr, nullptr, act4,
             MM, FFF, EE, FFF, FFF, 1, 0, 0, 1);
        // MLP fc2 + residual -> x
        conv_bf16(stream, W2 + (size_t)l * FFF * EE, wbuf, (size_t)FFF * EE);
        gemm(stream, act4, wbuf, b2w + (size_t)l * EE, x, x, nullptr,
             MM, EE, FFF, EE, EE, 1, 0, 0, 0);
    }
    // Final LN + LM head -> logits (f32, directly into d_out)
    layernorm_kernel<<<dim3(MM), dim3(256), 0, stream>>>(x, lnf_s, lnf_b, hbuf);
    conv_bf16(stream, Whead, wbuf, (size_t)EE * VV);
    gemm(stream, hbuf, wbuf, bhead, nullptr, logits, nullptr,
         MM, VV, EE, VV, VV, 1, 0, 0, 0);
    // Weighted CE loss
    loss_row_kernel<<<dim3(MM), dim3(256), 0, stream>>>(logits, targets, rloss);
    loss_reduce_kernel<<<dim3(1), dim3(256), 0, stream>>>(rloss, lossp, MM);
}
